// ProposalModule_57844619543183
// MI455X (gfx1250) — compile-verified, matrix operand order checked by reading, not run on hardware
//
#include <hip/hip_runtime.h>
#include <hip/hip_bf16.h>
#include <cstdint>

typedef _Float16 v16h __attribute__((ext_vector_type(16)));
typedef _Float16 v8h  __attribute__((ext_vector_type(8)));
typedef float    v8f  __attribute__((ext_vector_type(8)));

#define B_      32
#define K_      1024
#define NPROP_  256
#define NSAMP_  16
#define RAD_    0.3f
#define RAD2_   0.09f
#define OUTCH_  97
#define PI_     3.14159265358979323846f

// ---------------- FPS: one block per point cloud ----------------
__global__ void fps_kernel(const float* __restrict__ xyz, int* __restrict__ fidx,
                           float* __restrict__ newxyz) {
  __shared__ float sx[K_], sy[K_], sz[K_];
  __shared__ float rv[256];
  __shared__ int   ri[256];
  int b = blockIdx.x, t = threadIdx.x;
  const float* base = xyz + (size_t)b * K_ * 3;
  for (int j = 0; j < 4; ++j) {
    int p = j * 256 + t;
    sx[p] = base[p*3+0]; sy[p] = base[p*3+1]; sz[p] = base[p*3+2];
  }
  float dist[4] = {1e10f, 1e10f, 1e10f, 1e10f};
  __syncthreads();
  int far = 0;
  for (int i = 0; i < NPROP_; ++i) {
    if (t == 0) {
      fidx[b*NPROP_ + i] = far;
      newxyz[((size_t)b*NPROP_ + i)*3 + 0] = sx[far];
      newxyz[((size_t)b*NPROP_ + i)*3 + 1] = sy[far];
      newxyz[((size_t)b*NPROP_ + i)*3 + 2] = sz[far];
    }
    float fx = sx[far], fy = sy[far], fz = sz[far];
    float bv = -1.0f; int bidx = 0x7fffffff;
    #pragma unroll
    for (int j = 0; j < 4; ++j) {
      int p = j * 256 + t;
      float dx = sx[p]-fx, dy = sy[p]-fy, dz = sz[p]-fz;
      float d = dx*dx + dy*dy + dz*dz;
      dist[j] = dist[j] < d ? dist[j] : d;
      if (dist[j] > bv || (dist[j] == bv && p < bidx)) { bv = dist[j]; bidx = p; }
    }
    rv[t] = bv; ri[t] = bidx;
    __syncthreads();
    for (int s = 128; s > 0; s >>= 1) {
      if (t < s) {
        float v2 = rv[t+s]; int i2 = ri[t+s];
        if (v2 > rv[t] || (v2 == rv[t] && i2 < ri[t])) { rv[t] = v2; ri[t] = i2; }
      }
      __syncthreads();
    }
    far = ri[0];
    __syncthreads();
  }
}

// ------- ball query (first 16 by index within radius, wave32 ballot scan)
//         + gather into f16 layer-1 input (zero-padded to kpad) -------
__global__ void group_kernel(const float* __restrict__ xyz, const float* __restrict__ feats,
                             int Cf, const int* __restrict__ fidx,
                             const float* __restrict__ newxyz,
                             _Float16* __restrict__ X0, int kpad) {
  __shared__ int sidx[NSAMP_];
  int bp = blockIdx.x;
  int b = bp >> 8;
  int tid = threadIdx.x;
  float cx = newxyz[(size_t)bp*3 + 0];
  float cy = newxyz[(size_t)bp*3 + 1];
  float cz = newxyz[(size_t)bp*3 + 2];
  const float* xb = xyz + (size_t)b * K_ * 3;
  if (tid < 32) {
    int lane = tid;
    int count = 0;
    for (int base = 0; base < K_; base += 32) {
      int k = base + lane;
      float dx = xb[k*3+0]-cx, dy = xb[k*3+1]-cy, dz = xb[k*3+2]-cz;
      bool pred = (dx*dx + dy*dy + dz*dz) < RAD2_;
      unsigned long long mask = __ballot(pred ? 1 : 0);
      int pos = count + __popcll(mask & ((1ull << lane) - 1ull));
      if (pred && pos < NSAMP_) sidx[pos] = k;
      count += __popcll(mask);
      if (count >= NSAMP_) break;
    }
    if (lane == 0) {
      int c = count < NSAMP_ ? count : NSAMP_;
      int fill = (c > 0) ? sidx[0] : fidx[bp];
      for (int j = c; j < NSAMP_; ++j) sidx[j] = fill;
    }
  }
  __syncthreads();
  int tot = NSAMP_ * kpad;
  for (int e = tid; e < tot; e += blockDim.x) {
    int s = e / kpad, c = e - s * kpad;
    int k = sidx[s];
    float v;
    if (c < 3) {
      float cc = (c == 0) ? cx : ((c == 1) ? cy : cz);
      v = (xb[k*3+c] - cc) * (1.0f / RAD_);
    } else if (c < 3 + Cf) {
      v = feats[((size_t)b*Cf + (c-3))*K_ + k];
    } else {
      v = 0.0f;
    }
    X0[(size_t)(bp*NSAMP_ + s)*kpad + c] = (_Float16)v;
  }
}

// ---------------- weight f32 -> f16 (zero-padded 128 x kpad) ----------------
__global__ void prep_w(const float* __restrict__ W, _Float16* __restrict__ W16,
                       int co, int ci, int kpad) {
  int g = blockIdx.x * blockDim.x + threadIdx.x;
  if (g >= 128 * kpad) return;
  int o = g / kpad, k = g - o * kpad;
  float v = (o < co && k < ci) ? W[o*ci + k] : 0.0f;
  W16[g] = (_Float16)v;
}

// fold BN gamma / betas into a single per-output scale+bias epilogue
__global__ void prep_sb(const float* __restrict__ g, const float* __restrict__ b,
                        const float* __restrict__ be,
                        float* __restrict__ scale, float* __restrict__ bias,
                        int mode, int nvalid) {
  int o = threadIdx.x;
  if (o >= 128) return;
  const float invs = rsqrtf(1.0f + 1e-5f);
  if (mode == 0)      { float s = g[o]*invs; scale[o] = s; bias[o] = b[o]; }
  else if (mode == 1) { float s = g[o]*invs; scale[o] = s; bias[o] = b[o]*s + be[o]; }
  else                { scale[o] = 1.0f; bias[o] = (o < nvalid) ? b[o] : 0.0f; }
}

// -------- WMMA GEMM: Y[m,o] = act( (sum_k X[m,k]*W[o,k]) * scale[o] + bias[o] )
//          256 threads = 8 waves; wave = 16 rows x 128 cols; W staged in LDS --------
template<bool RELU, bool F32OUT>
__global__ void mlp_wmma(const _Float16* __restrict__ X, int ldX,
                         const _Float16* __restrict__ Wg, int kpad, int ksteps,
                         const float* __restrict__ scale, const float* __restrict__ bias,
                         void* __restrict__ Yv, int ldY, int M) {
  alignas(16) __shared__ _Float16 sW[128 * 160];
  __shared__ float sSc[128], sBi[128];
  int tid = threadIdx.x;
  for (int i = tid; i < 128 * kpad; i += 256) sW[i] = Wg[i];
  if (tid < 128) { sSc[tid] = scale[tid]; sBi[tid] = bias[tid]; }
  __syncthreads();

  int wave = tid >> 5, lane = tid & 31;
  int half = lane >> 4, lr = lane & 15;
  int m0 = blockIdx.x * 128 + wave * 16;   // M is always a multiple of 128

  v8f acc[8];
  #pragma unroll
  for (int nt = 0; nt < 8; ++nt)
    #pragma unroll
    for (int r = 0; r < 8; ++r) acc[nt][r] = 0.0f;

  const _Float16* xrow = X + (size_t)(m0 + lr) * ldX;
  for (int kk = 0; kk < ksteps; ++kk) {
    int kb = kk * 32 + half * 8;           // lane-half K mapping: {0..7,16..23}/{8..15,24..31}
    v8h alo = *(const v8h*)(xrow + kb);
    v8h ahi = *(const v8h*)(xrow + kb + 16);
    v16h a;
    #pragma unroll
    for (int i = 0; i < 8; ++i) { a[i] = alo[i]; a[i+8] = ahi[i]; }
    #pragma unroll
    for (int nt = 0; nt < 8; ++nt) {
      const _Float16* wrow = sW + (size_t)(nt*16 + lr) * kpad;
      v8h blo = *(const v8h*)(wrow + kb);
      v8h bhi = *(const v8h*)(wrow + kb + 16);
      v16h bb;
      #pragma unroll
      for (int i = 0; i < 8; ++i) { bb[i] = blo[i]; bb[i+8] = bhi[i]; }
      acc[nt] = __builtin_amdgcn_wmma_f32_16x16x32_f16(
          false, a, false, bb, (short)0, acc[nt], false, false);
    }
  }

  #pragma unroll
  for (int nt = 0; nt < 8; ++nt) {
    int o = nt*16 + lr;
    float sc = sSc[o], bi = sBi[o];
    #pragma unroll
    for (int r = 0; r < 8; ++r) {
      int m = m0 + r + half * 8;           // D: VGPR r -> M=r(+8 for upper lanes), N=lane%16
      float v = acc[nt][r] * sc + bi;
      if (RELU) v = v > 0.0f ? v : 0.0f;
      if (F32OUT) ((float*)Yv)[(size_t)m * ldY + o] = v;
      else        ((_Float16*)Yv)[(size_t)m * ldY + o] = (_Float16)v;
    }
  }
}

// ---------------- max over S=16 samples ----------------
__global__ void maxpool_kernel(const _Float16* __restrict__ Y, _Float16* __restrict__ P,
                               int ldY) {
  int g = blockIdx.x * blockDim.x + threadIdx.x;   // B*P*128 threads
  int bp = g >> 7, o = g & 127;
  const _Float16* src = Y + (size_t)(bp * NSAMP_) * ldY + o;
  float m = (float)src[0];
  for (int s = 1; s < NSAMP_; ++s) {
    float v = (float)src[(size_t)s * ldY];
    m = v > m ? v : m;
  }
  P[(size_t)bp * 128 + o] = (_Float16)m;
}

// ---------------- decode ----------------
__global__ void decode_kernel(const float* __restrict__ H, const float* __restrict__ nxyz,
                              const float* __restrict__ msize, float* __restrict__ out,
                              int branch) {
  int bp = blockIdx.x, ch = threadIdx.x;
  if (ch >= OUTCH_) return;
  float t = H[(size_t)bp * 128 + ch];
  float v;
  if (ch < 2)        v = t;                                   // objectness
  else if (ch < 5)   v = nxyz[(size_t)bp*3 + (ch-2)] + t;     // center
  else if (ch == 5)  v = t;                                   // heading score
  else if (ch == 6)  v = t * PI_;                             // heading residual (pi/NH, NH=1)
  else if (ch < 25)  v = t;                                   // size scores
  else if (ch < 79)  v = t * msize[ch - 25];                  // size residuals * mean_size
  else               v = t;                                   // semantic
  out[((size_t)branch * (B_*NPROP_) + bp) * OUTCH_ + ch] = v;
}

extern "C" void kernel_launch(void* const* d_in, const int* in_sizes, int n_in,
                              void* d_out, int out_size, void* d_ws, size_t ws_size,
                              hipStream_t stream) {
  (void)in_sizes; (void)n_in; (void)out_size; (void)ws_size;
  char* ws = (char*)d_ws;
  constexpr size_t OFF_A   = 0;                       // X0 (131072x160 f16) / Y2 reuse
  constexpr size_t OFF_Bf  = 41943040;                // Y1 / Y3 (131072x128 f16)
  constexpr size_t OFF_W16 = OFF_Bf + 33554432;       // 128x160 f16 weights
  constexpr size_t OFF_SC  = OFF_W16 + 40960;
  constexpr size_t OFF_BI  = OFF_SC + 512;
  constexpr size_t OFF_FI  = OFF_BI + 512;            // fps indices
  constexpr size_t OFF_NX  = OFF_FI + 32768;          // new_xyz
  constexpr size_t OFF_PL  = OFF_NX + 98304;          // pooled / head2 out
  constexpr size_t OFF_H1  = OFF_PL + 2097152;        // head1 out
  constexpr size_t OFF_HO  = OFF_H1 + 2097152;        // head3 f32 out

  _Float16* bufA = (_Float16*)(ws + OFF_A);
  _Float16* bufB = (_Float16*)(ws + OFF_Bf);
  _Float16* w16  = (_Float16*)(ws + OFF_W16);
  float* sc   = (float*)(ws + OFF_SC);
  float* bi   = (float*)(ws + OFF_BI);
  int*   fidx = (int*)(ws + OFF_FI);
  float* nxyz = (float*)(ws + OFF_NX);
  _Float16* pool = (_Float16*)(ws + OFF_PL);
  _Float16* h1   = (_Float16*)(ws + OFF_H1);
  float* hout = (float*)(ws + OFF_HO);
  float* out  = (float*)d_out;
  const float* msize = (const float*)d_in[36];

  // pytree (sorted-key) flattening: heads at 6/16/26, mean_size 36, sa at 37/46/55
  struct Cfg { int xi, fi, Cf, sa, hd; };
  const Cfg cfg[3] = {
    {0, 1, 128, 46, 16},   // main
    {2, 3, 131, 37,  6},   // corner
    {4, 5, 128, 55, 26},   // plane
  };

  const int M1 = B_ * NPROP_ * NSAMP_;   // 131072
  const int M2 = B_ * NPROP_;            // 8192

  for (int br = 0; br < 3; ++br) {
    const Cfg& c = cfg[br];
    const float* xyz   = (const float*)d_in[c.xi];
    const float* feats = (const float*)d_in[c.fi];
    int Cin = 3 + c.Cf;

    fps_kernel<<<B_, 256, 0, stream>>>(xyz, fidx, nxyz);
    group_kernel<<<M2, 128, 0, stream>>>(xyz, feats, c.Cf, fidx, nxyz, bufA, 160);

    // SA layer 1 (K=131/134 padded to 160)
    prep_w<<<(128*160)/256, 256, 0, stream>>>((const float*)d_in[c.sa+0], w16, 128, Cin, 160);
    prep_sb<<<1, 128, 0, stream>>>((const float*)d_in[c.sa+2], (const float*)d_in[c.sa+1], nullptr, sc, bi, 0, 128);
    mlp_wmma<true,false><<<M1/128, 256, 0, stream>>>(bufA, 160, w16, 160, 5, sc, bi, bufB, 128, M1);
    // SA layer 2
    prep_w<<<64, 256, 0, stream>>>((const float*)d_in[c.sa+3], w16, 128, 128, 128);
    prep_sb<<<1, 128, 0, stream>>>((const float*)d_in[c.sa+5], (const float*)d_in[c.sa+4], nullptr, sc, bi, 0, 128);
    mlp_wmma<true,false><<<M1/128, 256, 0, stream>>>(bufB, 128, w16, 128, 4, sc, bi, bufA, 128, M1);
    // SA layer 3
    prep_w<<<64, 256, 0, stream>>>((const float*)d_in[c.sa+6], w16, 128, 128, 128);
    prep_sb<<<1, 128, 0, stream>>>((const float*)d_in[c.sa+8], (const float*)d_in[c.sa+7], nullptr, sc, bi, 0, 128);
    mlp_wmma<true,false><<<M1/128, 256, 0, stream>>>(bufA, 128, w16, 128, 4, sc, bi, bufB, 128, M1);

    maxpool_kernel<<<(M2*128)/256, 256, 0, stream>>>(bufB, pool, 128);

    // Head CBR 1
    prep_w<<<64, 256, 0, stream>>>((const float*)d_in[c.hd+0], w16, 128, 128, 128);
    prep_sb<<<1, 128, 0, stream>>>((const float*)d_in[c.hd+8], (const float*)d_in[c.hd+3], (const float*)d_in[c.hd+6], sc, bi, 1, 128);
    mlp_wmma<true,false><<<M2/128, 256, 0, stream>>>(pool, 128, w16, 128, 4, sc, bi, h1, 128, M2);
    // Head CBR 2
    prep_w<<<64, 256, 0, stream>>>((const float*)d_in[c.hd+1], w16, 128, 128, 128);
    prep_sb<<<1, 128, 0, stream>>>((const float*)d_in[c.hd+9], (const float*)d_in[c.hd+4], (const float*)d_in[c.hd+7], sc, bi, 1, 128);
    mlp_wmma<true,false><<<M2/128, 256, 0, stream>>>(h1, 128, w16, 128, 4, sc, bi, pool, 128, M2);
    // Head final linear (97 outputs, padded weights, f32 out, no relu)
    prep_w<<<64, 256, 0, stream>>>((const float*)d_in[c.hd+2], w16, 97, 128, 128);
    prep_sb<<<1, 128, 0, stream>>>(nullptr, (const float*)d_in[c.hd+5], nullptr, sc, bi, 2, 97);
    mlp_wmma<false,true><<<M2/128, 256, 0, stream>>>(pool, 128, w16, 128, 4, sc, bi, hout, 128, M2);

    decode_kernel<<<M2, 128, 0, stream>>>(hout, nxyz, msize, out, br);
  }
}
